// BiLSTMModel_2808908611660
// MI455X (gfx1250) — matrix operand order repaired and checked by reading, not compiled, not run on hardware
//
#include <hip/hip_runtime.h>
#include <hip/hip_bf16.h>

typedef __attribute__((ext_vector_type(16))) __bf16 bf16x16;
typedef __attribute__((ext_vector_type(8)))  __bf16 bf16x8;
typedef __attribute__((ext_vector_type(8)))  float  f32x8;

#define B_TOT 1024
#define T_LEN 256
#define H_DIM 64
#define MT    64     // batch rows per block

// ---- fast transcendentals: CDNA5 has single-instruction V_TANH_F32 ----
#if __has_builtin(__builtin_amdgcn_tanhf)
__device__ __forceinline__ float ftanh(float x) { return __builtin_amdgcn_tanhf(x); }
#elif __has_builtin(__builtin_amdgcn_tanh_f32)
__device__ __forceinline__ float ftanh(float x) { return __builtin_amdgcn_tanh_f32(x); }
#else
__device__ __forceinline__ float ftanh(float x) {
  // tanh(x) = 1 - 2/(exp(2x)+1), with raw v_rcp_f32 (no IEEE div expansion)
  float e = __expf(2.0f * x);
  return 1.0f - 2.0f * __builtin_amdgcn_rcpf(e + 1.0f);
}
#endif
__device__ __forceinline__ float sigm(float x) { return __builtin_fmaf(ftanh(0.5f * x), 0.5f, 0.5f); }

__device__ __forceinline__ f32x8 wmma_bf16(bf16x16 a, bf16x16 b, f32x8 c) {
  return __builtin_amdgcn_wmma_f32_16x16x32_bf16(false, a, false, b, (short)0, c, false, false);
}

union AV { bf16x16 v; bf16x8 h[2]; };

// A tile 16x32 (M x K) bf16 from row-major memory with element stride.
// Lane l (0..15): row M=l, K 0..7 then 16..23 ; lane l+16: row M=l, K 8..15 then 24..31.
__device__ __forceinline__ bf16x16 load_a(const __bf16* base, int stride, int lane) {
  int row = lane & 15;
  int cb  = (lane >> 4) * 8;
  AV u;
  u.h[0] = *(const bf16x8*)(base + row * stride + cb);
  u.h[1] = *(const bf16x8*)(base + row * stride + cb + 16);
  return u.v;
}

// B tile 32x16 (K x N) from weights stored [N][K] row-major (natural PyTorch layout).
// Lane l (0..15): col n, K = k0..k0+7 then k0+16..k0+23 ; lane l+16: col n, K = k0+8..k0+15 then k0+24..k0+31.
__device__ __forceinline__ bf16x16 load_b(const __bf16* W, int K, int n0, int k0, int lane) {
  int n = n0 + (lane & 15);
  int k = k0 + (lane >> 4) * 8;
  AV u;
  u.h[0] = *(const bf16x8*)(W + n * K + k);
  u.h[1] = *(const bf16x8*)(W + n * K + k + 16);
  return u.v;
}

// ---------------- prep kernels ----------------
__global__ void cvt_bf16(const float* __restrict__ s, __bf16* __restrict__ d, int n) {
  int i = blockIdx.x * blockDim.x + threadIdx.x;
  if (i < n) d[i] = (__bf16)s[i];
}
__global__ void pad_wih0(const float* __restrict__ s, __bf16* __restrict__ d) { // 256x5 -> 256x32 (zero pad)
  int i = blockIdx.x * blockDim.x + threadIdx.x;
  if (i < 256 * 32) {
    int n = i >> 5, k = i & 31;
    d[i] = (__bf16)((k < 5) ? s[n * 5 + k] : 0.0f);
  }
}
__global__ void bias_add(const float* __restrict__ a, const float* __restrict__ b,
                         float* __restrict__ d, int n) {
  int i = blockIdx.x * blockDim.x + threadIdx.x;
  if (i < n) d[i] = a[i] + b[i];
}

// ---------------- layer 0: fwd+bwd persistent scan ----------------
// grid.x = 32 : blockIdx/16 = dir, blockIdx%16 = batch tile (64 rows)
__global__ __launch_bounds__(256) void lstm_l0(
    const float*  __restrict__ x,     // [B,T,5] fp32
    const __bf16* __restrict__ wih,   // [2][256][32] padded bf16
    const __bf16* __restrict__ whh,   // [2][256][64] bf16
    const float*  __restrict__ bias,  // [2][256] (b_ih+b_hh)
    __bf16*       __restrict__ l0out) // [B,T,128] bf16
{
  __shared__ __bf16 lds_x[64 * 48];   // 96B row stride (16B aligned)
  __shared__ __bf16 lds_h[64 * 80];   // 160B row stride

  int tid  = threadIdx.x;
  int lane = tid & 31, wave = tid >> 5;
  int qc = wave & 3, mh = wave >> 2;       // hidden chunk / M half
  int dir = blockIdx.x >> 4;
  int b0  = (blockIdx.x & 15) * MT;

  const __bf16* Wih = wih + dir * 256 * 32;
  const __bf16* Whh = whh + dir * 256 * 64;
  const float*  bs  = bias + dir * 256;

  for (int i = tid; i < 64 * 48; i += 256) lds_x[i] = (__bf16)0.0f;
  for (int i = tid; i < 64 * 80; i += 256) lds_h[i] = (__bf16)0.0f;

  float creg[2][8];
  #pragma unroll
  for (int mt = 0; mt < 2; ++mt)
    #pragma unroll
    for (int r = 0; r < 8; ++r) creg[mt][r] = 0.0f;

  int nbase[4];
  #pragma unroll
  for (int g = 0; g < 4; ++g) nbase[g] = g * 64 + qc * 16; // i,f,g,o column bases
  float bv[4];
  #pragma unroll
  for (int g = 0; g < 4; ++g) bv[g] = bs[nbase[g] + (lane & 15)];

  int j        = qc * 16 + (lane & 15);          // hidden unit owned by this lane
  int mrow_off = mh * 32 + ((lane >> 4) << 3);   // C-layout M base for this lane half

  for (int step = 0; step < T_LEN; ++step) {
    int t = dir ? (T_LEN - 1 - step) : step;
    __syncthreads();                             // prev h writes + prev x reads done
    for (int i = tid; i < 64 * 5; i += 256) {    // stage x_t (zero-padded region untouched)
      int m = i / 5, d = i - m * 5;
      lds_x[m * 48 + d] = (__bf16)x[((size_t)(b0 + m) * T_LEN + t) * 5 + d];
    }
    __syncthreads();                             // x staged

    bf16x16 Ax[2], Ah[2][2];
    #pragma unroll
    for (int mt = 0; mt < 2; ++mt) {
      int rb = mh * 32 + mt * 16;
      Ax[mt] = load_a(lds_x + rb * 48, 48, lane);
      #pragma unroll
      for (int kc = 0; kc < 2; ++kc)
        Ah[mt][kc] = load_a(lds_h + rb * 80 + kc * 32, 80, lane);
    }
    __syncthreads();                             // lds_h / lds_x reads complete

    f32x8 acc[4][2];
    #pragma unroll
    for (int g = 0; g < 4; ++g) {
      bf16x16 Bx  = load_b(Wih, 32, nbase[g], 0,  lane);
      bf16x16 Bh0 = load_b(Whh, 64, nbase[g], 0,  lane);
      bf16x16 Bh1 = load_b(Whh, 64, nbase[g], 32, lane);
      #pragma unroll
      for (int mt = 0; mt < 2; ++mt) {
        f32x8 c;
        #pragma unroll
        for (int r = 0; r < 8; ++r) c[r] = bv[g];
        c = wmma_bf16(Ax[mt],    Bx,  c);
        c = wmma_bf16(Ah[mt][0], Bh0, c);
        c = wmma_bf16(Ah[mt][1], Bh1, c);
        acc[g][mt] = c;
      }
    }

    #pragma unroll
    for (int mt = 0; mt < 2; ++mt) {
      #pragma unroll
      for (int r = 0; r < 8; ++r) {
        float iv = acc[0][mt][r], fv = acc[1][mt][r];
        float gv = acc[2][mt][r], ov = acc[3][mt][r];
        float cn = sigm(fv) * creg[mt][r] + sigm(iv) * ftanh(gv);
        float hv = sigm(ov) * ftanh(cn);
        creg[mt][r] = cn;
        int m = mrow_off + mt * 16 + r;
        __bf16 hb = (__bf16)hv;
        lds_h[m * 80 + j] = hb;
        l0out[((size_t)(b0 + m) * T_LEN + t) * 128 + dir * 64 + j] = hb;
      }
    }
  }
}

// ---------------- layer 1 forward scan (only final h needed) ----------------
__global__ __launch_bounds__(256) void lstm_l1_fwd(
    const __bf16* __restrict__ l0out, // [B,T,128]
    const __bf16* __restrict__ wih1,  // [256][128]
    const __bf16* __restrict__ whh1,  // [256][64]
    const float*  __restrict__ bias,  // [256]
    __bf16*       __restrict__ h1f)   // [B,64]
{
  __shared__ __bf16 lds_h[64 * 80];

  int tid  = threadIdx.x;
  int lane = tid & 31, wave = tid >> 5;
  int qc = wave & 3, mh = wave >> 2;
  int b0 = blockIdx.x * MT;

  for (int i = tid; i < 64 * 80; i += 256) lds_h[i] = (__bf16)0.0f;

  float creg[2][8];
  #pragma unroll
  for (int mt = 0; mt < 2; ++mt)
    #pragma unroll
    for (int r = 0; r < 8; ++r) creg[mt][r] = 0.0f;

  int nbase[4];
  #pragma unroll
  for (int g = 0; g < 4; ++g) nbase[g] = g * 64 + qc * 16;
  float bv[4];
  #pragma unroll
  for (int g = 0; g < 4; ++g) bv[g] = bias[nbase[g] + (lane & 15)];

  int j        = qc * 16 + (lane & 15);
  int mrow_off = mh * 32 + ((lane >> 4) << 3);

  for (int t = 0; t < T_LEN; ++t) {
    f32x8 acc[4][2];
    #pragma unroll
    for (int g = 0; g < 4; ++g)
      #pragma unroll
      for (int mt = 0; mt < 2; ++mt)
        #pragma unroll
        for (int r = 0; r < 8; ++r) acc[g][mt][r] = bv[g];

    __syncthreads();                    // prev-step h visible
    // recurrent part: K = 64 from LDS
    #pragma unroll
    for (int kc = 0; kc < 2; ++kc) {
      bf16x16 Ah[2];
      #pragma unroll
      for (int mt = 0; mt < 2; ++mt)
        Ah[mt] = load_a(lds_h + (mh * 32 + mt * 16) * 80 + kc * 32, 80, lane);
      #pragma unroll
      for (int g = 0; g < 4; ++g) {
        bf16x16 Bh = load_b(whh1, 64, nbase[g], kc * 32, lane);
        #pragma unroll
        for (int mt = 0; mt < 2; ++mt) acc[g][mt] = wmma_bf16(Ah[mt], Bh, acc[g][mt]);
      }
    }
    __syncthreads();                    // all lds_h reads done before overwrite

    // input part: K = 128 from global (layer-0 output, bf16)
    #pragma unroll
    for (int kc = 0; kc < 4; ++kc) {
      bf16x16 Axv[2];
      #pragma unroll
      for (int mt = 0; mt < 2; ++mt) {
        const __bf16* xb = l0out + ((size_t)(b0 + mh * 32 + mt * 16) * T_LEN + t) * 128 + kc * 32;
        Axv[mt] = load_a(xb, T_LEN * 128, lane);
      }
      #pragma unroll
      for (int g = 0; g < 4; ++g) {
        bf16x16 Bx = load_b(wih1, 128, nbase[g], kc * 32, lane);
        #pragma unroll
        for (int mt = 0; mt < 2; ++mt) acc[g][mt] = wmma_bf16(Axv[mt], Bx, acc[g][mt]);
      }
    }

    #pragma unroll
    for (int mt = 0; mt < 2; ++mt) {
      #pragma unroll
      for (int r = 0; r < 8; ++r) {
        float iv = acc[0][mt][r], fv = acc[1][mt][r];
        float gv = acc[2][mt][r], ov = acc[3][mt][r];
        float cn = sigm(fv) * creg[mt][r] + sigm(iv) * ftanh(gv);
        float hv = sigm(ov) * ftanh(cn);
        creg[mt][r] = cn;
        int m = mrow_off + mt * 16 + r;
        lds_h[m * 80 + j] = (__bf16)hv;
      }
    }
  }
  __syncthreads();
  for (int i = tid; i < 64 * 64; i += 256) {
    int m = i >> 6, jj = i & 63;
    h1f[(size_t)(b0 + m) * 64 + jj] = lds_h[m * 80 + jj];
  }
}

// ---------------- layer 1 backward: exactly one step (t = T-1, zero state) ----------------
__global__ __launch_bounds__(256) void lstm_l1_bwd_last(
    const __bf16* __restrict__ l0out,
    const __bf16* __restrict__ wih1r, // [256][128]
    const float*  __restrict__ bias,  // [256]
    __bf16*       __restrict__ h1r)   // [B,64]
{
  int tid  = threadIdx.x;
  int lane = tid & 31, wave = tid >> 5;
  int qc = wave & 3, mh = wave >> 2;
  int b0 = blockIdx.x * MT;
  const int t = T_LEN - 1;

  int nbase[4];
  #pragma unroll
  for (int g = 0; g < 4; ++g) nbase[g] = g * 64 + qc * 16;

  f32x8 acc[4][2];
  #pragma unroll
  for (int g = 0; g < 4; ++g) {
    float bvv = bias[nbase[g] + (lane & 15)];
    #pragma unroll
    for (int mt = 0; mt < 2; ++mt)
      #pragma unroll
      for (int r = 0; r < 8; ++r) acc[g][mt][r] = bvv;
  }

  #pragma unroll
  for (int kc = 0; kc < 4; ++kc) {
    bf16x16 Axv[2];
    #pragma unroll
    for (int mt = 0; mt < 2; ++mt) {
      const __bf16* xb = l0out + ((size_t)(b0 + mh * 32 + mt * 16) * T_LEN + t) * 128 + kc * 32;
      Axv[mt] = load_a(xb, T_LEN * 128, lane);
    }
    #pragma unroll
    for (int g = 0; g < 4; ++g) {
      bf16x16 Bx = load_b(wih1r, 128, nbase[g], kc * 32, lane);
      #pragma unroll
      for (int mt = 0; mt < 2; ++mt) acc[g][mt] = wmma_bf16(Axv[mt], Bx, acc[g][mt]);
    }
  }

  int j        = qc * 16 + (lane & 15);
  int mrow_off = mh * 32 + ((lane >> 4) << 3);
  #pragma unroll
  for (int mt = 0; mt < 2; ++mt) {
    #pragma unroll
    for (int r = 0; r < 8; ++r) {
      float iv = acc[0][mt][r], gv = acc[2][mt][r], ov = acc[3][mt][r];
      float cn = sigm(iv) * ftanh(gv);            // c_prev = 0 -> forget term vanishes
      float hv = sigm(ov) * ftanh(cn);
      int m = mrow_off + mt * 16 + r;
      h1r[(size_t)(b0 + m) * 64 + j] = (__bf16)hv;
    }
  }
}

// ---------------- final FC ----------------
__global__ void fc_kernel(const __bf16* __restrict__ h1f, const __bf16* __restrict__ h1r,
                          const float* __restrict__ fc_w, const float* __restrict__ fc_b,
                          float* __restrict__ out) {
  int b = blockIdx.x * blockDim.x + threadIdx.x;
  if (b >= B_TOT) return;
  float acc = fc_b[0];
  #pragma unroll 8
  for (int jj = 0; jj < 64; ++jj) {
    acc += (float)h1f[(size_t)b * 64 + jj] * fc_w[jj];
    acc += (float)h1r[(size_t)b * 64 + jj] * fc_w[64 + jj];
  }
  out[b] = acc;
}

extern "C" void kernel_launch(void* const* d_in, const int* in_sizes, int n_in,
                              void* d_out, int out_size, void* d_ws, size_t ws_size,
                              hipStream_t stream) {
  (void)in_sizes; (void)n_in; (void)out_size; (void)ws_size;
  const float* x       = (const float*)d_in[0];
  const float* wih_l0  = (const float*)d_in[1];
  const float* whh_l0  = (const float*)d_in[2];
  const float* bih_l0  = (const float*)d_in[3];
  const float* bhh_l0  = (const float*)d_in[4];
  const float* wih_l0r = (const float*)d_in[5];
  const float* whh_l0r = (const float*)d_in[6];
  const float* bih_l0r = (const float*)d_in[7];
  const float* bhh_l0r = (const float*)d_in[8];
  const float* wih_l1  = (const float*)d_in[9];
  const float* whh_l1  = (const float*)d_in[10];
  const float* bih_l1  = (const float*)d_in[11];
  const float* bhh_l1  = (const float*)d_in[12];
  const float* wih_l1r = (const float*)d_in[13];
  const float* whh_l1r = (const float*)d_in[14];
  const float* bih_l1r = (const float*)d_in[15];
  const float* bhh_l1r = (const float*)d_in[16];
  const float* fc_w    = (const float*)d_in[17];
  const float* fc_b    = (const float*)d_in[18];

  char* ws = (char*)d_ws;
  size_t off = 0;
  __bf16* l0out = (__bf16*)(ws + off); off += (size_t)B_TOT * T_LEN * 128 * 2; // 67 MB
  __bf16* whhB  = (__bf16*)(ws + off); off += (size_t)4 * 256 * 64 * 2;        // l0,l0r,l1,l1r
  __bf16* wih1B = (__bf16*)(ws + off); off += (size_t)2 * 256 * 128 * 2;       // l1,l1r
  __bf16* wih0B = (__bf16*)(ws + off); off += (size_t)2 * 256 * 32 * 2;        // padded
  float*  biasS = (float*)(ws + off);  off += (size_t)4 * 256 * 4;
  __bf16* h1f   = (__bf16*)(ws + off); off += (size_t)B_TOT * 64 * 2;
  __bf16* h1r   = (__bf16*)(ws + off); off += (size_t)B_TOT * 64 * 2;

  cvt_bf16<<<64, 256, 0, stream>>>(whh_l0,  whhB + 0,      16384);
  cvt_bf16<<<64, 256, 0, stream>>>(whh_l0r, whhB + 16384,  16384);
  cvt_bf16<<<64, 256, 0, stream>>>(whh_l1,  whhB + 32768,  16384);
  cvt_bf16<<<64, 256, 0, stream>>>(whh_l1r, whhB + 49152,  16384);
  cvt_bf16<<<128, 256, 0, stream>>>(wih_l1,  wih1B + 0,     32768);
  cvt_bf16<<<128, 256, 0, stream>>>(wih_l1r, wih1B + 32768, 32768);
  pad_wih0<<<32, 256, 0, stream>>>(wih_l0,  wih0B + 0);
  pad_wih0<<<32, 256, 0, stream>>>(wih_l0r, wih0B + 8192);
  bias_add<<<1, 256, 0, stream>>>(bih_l0,  bhh_l0,  biasS + 0,   256);
  bias_add<<<1, 256, 0, stream>>>(bih_l0r, bhh_l0r, biasS + 256, 256);
  bias_add<<<1, 256, 0, stream>>>(bih_l1,  bhh_l1,  biasS + 512, 256);
  bias_add<<<1, 256, 0, stream>>>(bih_l1r, bhh_l1r, biasS + 768, 256);

  lstm_l0<<<32, 256, 0, stream>>>(x, wih0B, whhB, biasS, l0out);
  lstm_l1_fwd<<<16, 256, 0, stream>>>(l0out, wih1B, whhB + 32768, biasS + 512, h1f);
  lstm_l1_bwd_last<<<16, 256, 0, stream>>>(l0out, wih1B + 32768, biasS + 768, h1r);
  fc_kernel<<<4, 256, 0, stream>>>(h1f, h1r, fc_w, fc_b, (float*)d_out);
}
